// PPYOLOEHead_4913442587156
// MI455X (gfx1250) — matrix operand order repaired and checked by reading, not compiled, not run on hardware
//
#include <hip/hip_runtime.h>
#include <math.h>

// MI455X / gfx1250, wave32. WMMA f16->f32 path: v_wmma_f32_16x16x32_f16.
typedef __attribute__((ext_vector_type(16))) _Float16 v16h;
typedef __attribute__((ext_vector_type(8)))  _Float16 v8h;
typedef __attribute__((ext_vector_type(4)))  _Float16 v4h;
typedef __attribute__((ext_vector_type(8)))  float    v8f;

#define NUM_CLASSES 80
#define REG_CH      68     // 4*(REG_MAX+1)
#define MT          80     // padded M for pred convs (5 x 16)
#define TOTAL_L     8400
#define BATCH       16

__device__ __forceinline__ v16h cat8(v8h lo, v8h hi) {
  return __builtin_shufflevector(lo, hi, 0,1,2,3,4,5,6,7,8,9,10,11,12,13,14,15);
}
// A fragment (16x32 f16): lane row=lan, half; elems 0..7 -> K=half*8+e, 8..15 -> K=16+half*8+e.
__device__ __forceinline__ v16h load_a_frag(const _Float16* rowp, int half) {
  v8h lo = *(const v8h*)(rowp + half * 8);
  v8h hi = *(const v8h*)(rowp + 16 + half * 8);
  return cat8(lo, hi);
}
// B fragment (32x16 f16): lane col n, half; elems 0..7 -> K=half*8+e, 8..15 -> K=16+half*8+e. Bs stored transposed.
__device__ __forceinline__ v16h load_b_frag(const _Float16* colp, int half) {
  v8h lo = *(const v8h*)(colp + half * 8);
  v8h hi = *(const v8h*)(colp + 16 + half * 8);
  return cat8(lo, hi);
}

// ----------------------------------------------------- weight preconversion ---
__global__ __launch_bounds__(256) void f32_to_f16_kernel(const float* __restrict__ src,
                                                         _Float16* __restrict__ dst, int n) {
  int i = blockIdx.x * 256 + threadIdx.x;
  if (i < n) dst[i] = (_Float16)src[i];
}
// Wp [M][C][3][3] -> dst [tap][MT][C] (k innermost, rows >= M zeroed)
__global__ __launch_bounds__(256) void wp_swizzle_kernel(const float* __restrict__ Wp,
                                                         _Float16* __restrict__ dst,
                                                         int C, int M) {
  int i = blockIdx.x * 256 + threadIdx.x;
  if (i >= 9 * MT * C) return;
  int tap = i / (MT * C);
  int rem = i - tap * MT * C;
  int m = rem / C, k = rem - m * C;
  dst[i] = (m < M) ? (_Float16)Wp[((size_t)m * C + k) * 9 + tap] : (_Float16)0.f;
}

// ---------------------------------------------------------------- stage A ---
__global__ __launch_bounds__(256) void avg_kernel(const float* __restrict__ feat,
                                                  float* __restrict__ avg,
                                                  int C, int L) {
  __shared__ float red[256];
  int bc = blockIdx.x;
  const float* p = feat + (size_t)bc * L;
  float s = 0.f;
  for (int i = threadIdx.x; i < L; i += 256) s += p[i];
  red[threadIdx.x] = s;
  __syncthreads();
  for (int off = 128; off > 0; off >>= 1) {
    if (threadIdx.x < off) red[threadIdx.x] += red[threadIdx.x + off];
    __syncthreads();
  }
  if (threadIdx.x == 0) avg[bc] = red[0] / (float)L;
}

__global__ __launch_bounds__(256) void gate_kernel(const float* __restrict__ avg,
                                                   const float* __restrict__ fcw_cls,
                                                   const float* __restrict__ fcb_cls,
                                                   const float* __restrict__ fcw_reg,
                                                   const float* __restrict__ fcb_reg,
                                                   float* __restrict__ gate_cls,
                                                   float* __restrict__ gate_reg,
                                                   int C) {
  int t = blockIdx.x * 256 + threadIdx.x;
  int per = BATCH * C;
  if (t >= 2 * per) return;
  int branch = t / per;
  int rem = t - branch * per;
  int b = rem / C, co = rem - b * C;
  const float* w  = branch ? fcw_reg : fcw_cls;
  const float* bb = branch ? fcb_reg : fcb_cls;
  const float* a  = avg + (size_t)b * C;
  const float* wr = w + (size_t)co * C;
  float s = bb[co];
  for (int ci = 0; ci < C; ++ci) s += wr[ci] * a[ci];
  (branch ? gate_reg : gate_cls)[rem] = 1.f / (1.f + __expf(-s));
}

// ---------------------------------------------------------------- stage B ---
// 1x1 conv on (feat*gate) as batched GEMM. BM=128, BN=128, BK=32.
// 8 waves (4x2); each wave 32(M)x64(N) = 2x4 WMMA frags -> 8 WMMA/tile.
#define ASTR  40
#define BSTRT 40
__global__ __launch_bounds__(256) void ese_gemm_kernel(const _Float16* __restrict__ Wch,
                                                       const float* __restrict__ feat,
                                                       const float* __restrict__ gate,
                                                       const float* __restrict__ bn_s,
                                                       const float* __restrict__ bn_b,
                                                       _Float16* __restrict__ outp,
                                                       int C, int L, int addFeat) {
  __shared__ _Float16 As[128 * ASTR];
  __shared__ _Float16 Bs[128 * BSTRT];
  int tid = threadIdx.x;
  int b   = blockIdx.z;
  int co0 = blockIdx.y * 128;
  int l0  = blockIdx.x * 128;
  int wave = tid >> 5, lane = tid & 31;
  int waveM = wave >> 1, waveN = wave & 1;
  int half = lane >> 4, lan = lane & 15;

  v8f acc[2][4] = {};
  const float* gb = gate + (size_t)b * C;
  const float* fb = feat + (size_t)b * C * L;

  for (int k0 = 0; k0 < C; k0 += 32) {
    // A tile: 128x32 f16, pure 16B copies (4 v8h groups per row)
    for (int i = tid; i < 128 * 4; i += 256) {
      int r = i >> 2, c8 = (i & 3) * 8;
      *(v8h*)&As[r * ASTR + c8] = *(const v8h*)&Wch[(size_t)(co0 + r) * C + (k0 + c8)];
    }
    // B tile: 32x128 of feat*gate, transposed store; float4 global reads
    for (int i = tid; i < 32 * 32; i += 256) {
      int kk = i >> 5, n4 = (i & 31) * 4;
      int l = l0 + n4;
      float g = gb[k0 + kk];
      float4 x = make_float4(0.f, 0.f, 0.f, 0.f);
      if (l < L) x = *(const float4*)&fb[(size_t)(k0 + kk) * L + l];  // L%4==0 -> all-or-none
      Bs[(n4 + 0) * BSTRT + kk] = (_Float16)(x.x * g);
      Bs[(n4 + 1) * BSTRT + kk] = (_Float16)(x.y * g);
      Bs[(n4 + 2) * BSTRT + kk] = (_Float16)(x.z * g);
      Bs[(n4 + 3) * BSTRT + kk] = (_Float16)(x.w * g);
    }
    __syncthreads();

    v16h aF[2], bF[4];
#pragma unroll
    for (int mb = 0; mb < 2; ++mb)
      aF[mb] = load_a_frag(&As[(waveM * 32 + mb * 16 + lan) * ASTR], half);
#pragma unroll
    for (int nb = 0; nb < 4; ++nb)
      bF[nb] = load_b_frag(&Bs[(waveN * 64 + nb * 16 + lan) * BSTRT], half);
#pragma unroll
    for (int mb = 0; mb < 2; ++mb)
#pragma unroll
      for (int nb = 0; nb < 4; ++nb)
        acc[mb][nb] = __builtin_amdgcn_wmma_f32_16x16x32_f16(
            false, aF[mb], false, bF[nb], (short)0, acc[mb][nb], false, false);
    __syncthreads();
  }

  // fused epilogue: BN affine + swish (+feat for cls branch); store f16
#pragma unroll
  for (int mb = 0; mb < 2; ++mb) {
#pragma unroll
    for (int nb = 0; nb < 4; ++nb) {
      int nc = l0 + waveN * 64 + nb * 16 + lan;
      if (nc >= L) continue;
#pragma unroll
      for (int r = 0; r < 8; ++r) {
        int m = co0 + waveM * 32 + mb * 16 + r + half * 8;
        float y  = acc[mb][nb][r] * bn_s[m] + bn_b[m];
        float sw = y * (1.f / (1.f + __expf(-y)));
        if (addFeat) sw += fb[(size_t)m * L + nc];
        outp[((size_t)b * C + m) * L + nc] = (_Float16)sw;
      }
    }
  }
}

// ---------------------------------------------------------------- stage C ---
// 3x3 pred conv as implicit GEMM: MT=80 rows (5 frags), K = 9*C, BN=256
// (8 waves x 32 cols = 2 N-frags each -> 10 WMMA per (5 A + 2 B) frag loads).
#define A3STR 40
#define B3STRT 40
__global__ __launch_bounds__(256) void conv3_wmma_kernel(const _Float16* __restrict__ Wph,
                                                         const float* __restrict__ bias,
                                                         const _Float16* __restrict__ inp,
                                                         int C, int Hd, int Wd, int M,
                                                         float* __restrict__ out_cls,
                                                         int scale_off,
                                                         float* __restrict__ out_raw,
                                                         int mode) {
  __shared__ _Float16 As[MT * A3STR];
  __shared__ _Float16 Bs[256 * B3STRT];
  int L = Hd * Wd;
  int tid = threadIdx.x;
  int b   = blockIdx.z;
  int l0  = blockIdx.x * 256;
  int wave = tid >> 5, lane = tid & 31;
  int half = lane >> 4, lan = lane & 15;
  const _Float16* inb = inp + (size_t)b * C * L;

  v8f acc[5][2] = {};

  for (int tap = 0; tap < 9; ++tap) {
    int dy = tap / 3 - 1, dx = tap % 3 - 1;
    const _Float16* wt = Wph + (size_t)tap * MT * C;
    for (int kc = 0; kc < C; kc += 32) {
      // A tile: 80x32 f16, pure 16B copies
      for (int i = tid; i < MT * 4; i += 256) {
        int m = i >> 2, c8 = (i & 3) * 8;
        *(v8h*)&As[m * A3STR + c8] = *(const v8h*)&wt[(size_t)m * C + (kc + c8)];
      }
      // B tile: 32x256 shifted input, transposed store
      if (dx == 0) {
        // fast path: 4-group of l never crosses a row (Wd%4==0)
        for (int i = tid; i < 32 * 64; i += 256) {
          int kk = i >> 6, n4 = (i & 63) * 4;
          int l = l0 + n4;
          v4h v = {};
          if (l < L) {
            int y = l / Wd, x = l - y * Wd;
            int yy = y + dy;
            if (yy >= 0 && yy < Hd)
              v = *(const v4h*)&inb[(size_t)(kc + kk) * L + yy * Wd + x];
          }
          Bs[(n4 + 0) * B3STRT + kk] = v[0];
          Bs[(n4 + 1) * B3STRT + kk] = v[1];
          Bs[(n4 + 2) * B3STRT + kk] = v[2];
          Bs[(n4 + 3) * B3STRT + kk] = v[3];
        }
      } else {
        for (int i = tid; i < 32 * 256; i += 256) {
          int kk = i >> 8, n = i & 255;
          int l = l0 + n;
          _Float16 v = (_Float16)0.f;
          if (l < L) {
            int y = l / Wd, x = l - y * Wd;
            int yy = y + dy, xx = x + dx;
            if (yy >= 0 && yy < Hd && xx >= 0 && xx < Wd)
              v = inb[(size_t)(kc + kk) * L + yy * Wd + xx];
          }
          Bs[n * B3STRT + kk] = v;
        }
      }
      __syncthreads();

      v16h bF[2];
#pragma unroll
      for (int nb = 0; nb < 2; ++nb)
        bF[nb] = load_b_frag(&Bs[(wave * 32 + nb * 16 + lan) * B3STRT], half);
#pragma unroll
      for (int f = 0; f < 5; ++f) {
        v16h aF = load_a_frag(&As[(f * 16 + lan) * A3STR], half);
#pragma unroll
        for (int nb = 0; nb < 2; ++nb)
          acc[f][nb] = __builtin_amdgcn_wmma_f32_16x16x32_f16(
              false, aF, false, bF[nb], (short)0, acc[f][nb], false, false);
      }
      __syncthreads();
    }
  }

#pragma unroll
  for (int nb = 0; nb < 2; ++nb) {
    int ncol = l0 + wave * 32 + nb * 16 + lan;
    if (ncol >= L) continue;
#pragma unroll
    for (int f = 0; f < 5; ++f) {
#pragma unroll
      for (int r = 0; r < 8; ++r) {
        int m = f * 16 + r + half * 8;
        if (m < M) {
          float v = acc[f][nb][r] + bias[m];
          if (mode == 0) {
            out_cls[((size_t)b * NUM_CLASSES + m) * TOTAL_L + scale_off + ncol] =
                1.f / (1.f + __expf(-v));
          } else {
            out_raw[((size_t)b * REG_CH + m) * L + ncol] = v;
          }
        }
      }
    }
  }
}

// ----------------------------------------------------- DFL softmax+integral ---
__global__ __launch_bounds__(256) void dfl_kernel(const float* __restrict__ raw,
                                                  float* __restrict__ outp,
                                                  int L, int scale_off) {
  int idx = blockIdx.x * 256 + threadIdx.x;
  if (idx >= BATCH * 4 * L) return;
  int b = idx / (4 * L);
  int rem = idx - b * 4 * L;
  int f = rem / L;
  int l = rem - f * L;
  const float* p = raw + ((size_t)b * REG_CH + f * 17) * L + l;
  float v[17];
  float mx = -1e30f;
#pragma unroll
  for (int r = 0; r < 17; ++r) { v[r] = p[(size_t)r * L]; mx = fmaxf(mx, v[r]); }
  float s = 0.f, d = 0.f;
#pragma unroll
  for (int r = 0; r < 17; ++r) { float e = __expf(v[r] - mx); s += e; d += e * (float)r; }
  outp[((size_t)b * 4 + f) * TOTAL_L + scale_off + l] = d / s;
}

// --------------------------------------------------------------- anchors ---
__global__ __launch_bounds__(256) void anchors_kernel(float* __restrict__ base) {
  int g = blockIdx.x * 256 + threadIdx.x;
  if (g >= TOTAL_L) return;
  int w, loc; float s;
  if (g < 400)       { w = 20; s = 32.f; loc = g; }
  else if (g < 2000) { w = 40; s = 16.f; loc = g - 400; }
  else               { w = 80; s = 8.f;  loc = g - 2000; }
  int y = loc / w, x = loc - y * w;
  base[g * 2 + 0] = (float)x + 0.5f;
  base[g * 2 + 1] = (float)y + 0.5f;
  base[TOTAL_L * 2 + g] = s;
}

// ---------------------------------------------------------------------------
extern "C" void kernel_launch(void* const* d_in, const int* in_sizes, int n_in,
                              void* d_out, int out_size, void* d_ws, size_t ws_size,
                              hipStream_t stream) {
  (void)in_sizes; (void)n_in; (void)out_size; (void)ws_size;
  float* out = (float*)d_out;
  char*  ws  = (char*)d_ws;

  const size_t CMAX = 1024;
  float* avg      = (float*)ws;                 // [16,1024]
  float* gate_cls = avg + BATCH * CMAX;
  float* gate_reg = gate_cls + BATCH * CMAX;
  size_t off = 3 * BATCH * CMAX * sizeof(float);
  off = (off + 255) & ~(size_t)255;
  const size_t eseElems = (size_t)BATCH * 1638400;   // max C*L (256*6400)
  _Float16* ese_cls = (_Float16*)(ws + off); off += eseElems * sizeof(_Float16);
  off = (off + 255) & ~(size_t)255;
  _Float16* ese_reg = (_Float16*)(ws + off); off += eseElems * sizeof(_Float16);
  off = (off + 255) & ~(size_t)255;
  float* reg_raw = (float*)(ws + off);               // [16,68,6400]
  off += (size_t)BATCH * REG_CH * 6400 * sizeof(float);
  off = (off + 255) & ~(size_t)255;
  _Float16* wch_cls = (_Float16*)(ws + off); off += CMAX * CMAX * sizeof(_Float16);
  _Float16* wch_reg = (_Float16*)(ws + off); off += CMAX * CMAX * sizeof(_Float16);
  _Float16* wph_cls = (_Float16*)(ws + off); off += (size_t)9 * MT * CMAX * sizeof(_Float16);
  _Float16* wph_reg = (_Float16*)(ws + off); off += (size_t)9 * MT * CMAX * sizeof(_Float16);

  const int Cs[3] = {1024, 512, 256};
  const int Hs[3] = {20, 40, 80};
  const int SO[3] = {0, 400, 2000};

  float* cls_out = out;                                           // [16,80,8400]
  float* reg_out = out + (size_t)BATCH * NUM_CLASSES * TOTAL_L;   // [16,4,8400]
  float* anch    = reg_out + (size_t)BATCH * 4 * TOTAL_L;         // [8400,2]+[8400]

  for (int i = 0; i < 3; ++i) {
    const int base = i * 15;
    const float* feat = (const float*)d_in[base + 0];
    const float* cfw  = (const float*)d_in[base + 1];
    const float* cfb  = (const float*)d_in[base + 2];
    const float* ccw  = (const float*)d_in[base + 3];
    const float* cbs  = (const float*)d_in[base + 4];
    const float* cbb  = (const float*)d_in[base + 5];
    const float* rfw  = (const float*)d_in[base + 6];
    const float* rfb  = (const float*)d_in[base + 7];
    const float* rcw  = (const float*)d_in[base + 8];
    const float* rbs  = (const float*)d_in[base + 9];
    const float* rbb  = (const float*)d_in[base + 10];
    const float* pcw  = (const float*)d_in[base + 11];
    const float* pcb  = (const float*)d_in[base + 12];
    const float* prw  = (const float*)d_in[base + 13];
    const float* prb  = (const float*)d_in[base + 14];
    int C = Cs[i], H = Hs[i], W = Hs[i], L = H * W;

    // weight preconversion (f32 -> f16; pred weights re-swizzled [tap][m][k])
    int ncc = C * C;
    f32_to_f16_kernel<<<(ncc + 255) / 256, 256, 0, stream>>>(ccw, wch_cls, ncc);
    f32_to_f16_kernel<<<(ncc + 255) / 256, 256, 0, stream>>>(rcw, wch_reg, ncc);
    int nwp = 9 * MT * C;
    wp_swizzle_kernel<<<(nwp + 255) / 256, 256, 0, stream>>>(pcw, wph_cls, C, NUM_CLASSES);
    wp_swizzle_kernel<<<(nwp + 255) / 256, 256, 0, stream>>>(prw, wph_reg, C, REG_CH);

    avg_kernel<<<BATCH * C, 256, 0, stream>>>(feat, avg, C, L);
    int gthreads = 2 * BATCH * C;
    gate_kernel<<<(gthreads + 255) / 256, 256, 0, stream>>>(
        avg, cfw, cfb, rfw, rfb, gate_cls, gate_reg, C);

    dim3 g1((L + 127) / 128, C / 128, BATCH);
    ese_gemm_kernel<<<g1, 256, 0, stream>>>(wch_cls, feat, gate_cls, cbs, cbb, ese_cls, C, L, 1);
    ese_gemm_kernel<<<g1, 256, 0, stream>>>(wch_reg, feat, gate_reg, rbs, rbb, ese_reg, C, L, 0);

    dim3 g2((L + 255) / 256, 1, BATCH);
    conv3_wmma_kernel<<<g2, 256, 0, stream>>>(wph_cls, pcb, ese_cls, C, H, W, NUM_CLASSES,
                                              cls_out, SO[i], nullptr, 0);
    conv3_wmma_kernel<<<g2, 256, 0, stream>>>(wph_reg, prb, ese_reg, C, H, W, REG_CH,
                                              nullptr, 0, reg_raw, 1);

    int dtot = BATCH * 4 * L;
    dfl_kernel<<<(dtot + 255) / 256, 256, 0, stream>>>(reg_raw, reg_out, L, SO[i]);
  }
  anchors_kernel<<<(TOTAL_L + 255) / 256, 256, 0, stream>>>(anch);
}
